// ResidualVectorQuantizer_17102559772723
// MI455X (gfx1250) — compile-verified, matrix operand order checked by reading, not run on hardware
//
#include <hip/hip_runtime.h>

typedef __attribute__((ext_vector_type(16))) __bf16 v16bf;
typedef __attribute__((ext_vector_type(8)))  __bf16 v8bf;
typedef __attribute__((ext_vector_type(8)))  float  v8f;
typedef __attribute__((ext_vector_type(4)))  int    v4i;

#define B_N     65536
#define D_DIM   384
#define K_CODES 1024
#define L_LEV   4
#define BETA_C  0.25f

#define CHUNK_U4 1536   // 32 codes x 384 dims x 2B / 16B = 1536 uint4 = 24 KB

#if defined(__gfx1250__) && \
    __has_builtin(__builtin_amdgcn_global_load_async_to_lds_b128) && \
    __has_builtin(__builtin_amdgcn_s_wait_asynccnt)
#define RVQ_ASYNC 1
typedef __attribute__((address_space(1))) v4i* rvq_gv4_t;   // global v4i*
typedef __attribute__((address_space(3))) v4i* rvq_lv4_t;   // LDS v4i*
#else
#define RVQ_ASYNC 0
#endif

// ---------------------------------------------------------------------------
// Prep: bf16 codebook copy + per-code squared norms + zero the loss accumulator
// one wave (32 lanes) per code; 4096 codes total
// ---------------------------------------------------------------------------
__global__ __launch_bounds__(256) void rvq_prep_kernel(
    const float* __restrict__ cb, __bf16* __restrict__ cb_bf,
    float* __restrict__ norms, float* __restrict__ lossAcc) {
  int wid  = blockIdx.x * 8 + (threadIdx.x >> 5);   // code id in [0, L*K)
  int lane = threadIdx.x & 31;
  const float* row = cb + (size_t)wid * D_DIM;
  __bf16* orow = cb_bf + (size_t)wid * D_DIM;
  float s = 0.f;
#pragma unroll
  for (int t = 0; t < 12; ++t) {
    int d = lane + 32 * t;
    float v = row[d];
    s += v * v;
    orow[d] = (__bf16)v;
  }
#pragma unroll
  for (int off = 16; off >= 1; off >>= 1) s += __shfl_xor(s, off, 32);
  if (lane == 0) norms[wid] = s;
  if (blockIdx.x == 0 && threadIdx.x == 0) *lossAcc = 0.f;
}

// ---------------------------------------------------------------------------
// Async staging of one 24 KB codebook chunk (6 x b128 per thread, 256 threads)
// ---------------------------------------------------------------------------
__device__ __forceinline__ void rvq_stage_async(const uint4* gsrc, uint4* ldst,
                                                int tid) {
#if RVQ_ASYNC
#pragma unroll
  for (int i = 0; i < 6; ++i) {
    int e = tid + i * 256;
    __builtin_amdgcn_global_load_async_to_lds_b128(
        (rvq_gv4_t)(gsrc + e), (rvq_lv4_t)(ldst + e), 0, 0);
  }
#endif
}

// ---------------------------------------------------------------------------
// One VQ level: WMMA distance scores + argmin + gather + residual update + loss
// 256 threads = 8 waves; each wave owns 16 rows; block owns 128 rows
// ---------------------------------------------------------------------------
__global__ __launch_bounds__(256) void rvq_level_kernel(
    const float* __restrict__ resIn, float* __restrict__ resOut,
    const float* __restrict__ cb, const __bf16* __restrict__ cb_bf,
    const float* __restrict__ norms, float* __restrict__ out,
    float* __restrict__ lossAcc, int level) {

  __shared__ uint4 ldsCB[2 * CHUNK_U4];   // 48 KB double buffer

  const int tid   = threadIdx.x;
  const int lane  = tid & 31;
  const int wave  = tid >> 5;
  const int mrow  = lane & 15;            // row (A) / code-col (B) within tile
  const int kbase = (lane >> 4) * 8;      // ISA A/B lane->K mapping base
  const int rowBase = blockIdx.x * 128 + wave * 16;

  // ---- load A fragments: 16 rows x 384 dims, fp32 -> bf16, WMMA lane layout
  v16bf a[12];
  const float* rrow = resIn + (size_t)(rowBase + mrow) * D_DIM;
#pragma unroll
  for (int c = 0; c < 12; ++c) {
    const int d0 = c * 32 + kbase;
    float4 f0 = *(const float4*)(rrow + d0);        // K = d0..d0+3
    float4 f1 = *(const float4*)(rrow + d0 + 4);    // K = d0+4..d0+7
    float4 f2 = *(const float4*)(rrow + d0 + 16);   // K = d0+16..d0+19
    float4 f3 = *(const float4*)(rrow + d0 + 20);   // K = d0+20..d0+23
    a[c][0]  = (__bf16)f0.x;  a[c][1]  = (__bf16)f0.y;
    a[c][2]  = (__bf16)f0.z;  a[c][3]  = (__bf16)f0.w;
    a[c][4]  = (__bf16)f1.x;  a[c][5]  = (__bf16)f1.y;
    a[c][6]  = (__bf16)f1.z;  a[c][7]  = (__bf16)f1.w;
    a[c][8]  = (__bf16)f2.x;  a[c][9]  = (__bf16)f2.y;
    a[c][10] = (__bf16)f2.z;  a[c][11] = (__bf16)f2.w;
    a[c][12] = (__bf16)f3.x;  a[c][13] = (__bf16)f3.y;
    a[c][14] = (__bf16)f3.z;  a[c][15] = (__bf16)f3.w;
  }

  float minval[8];
  int   minidx[8];
#pragma unroll
  for (int j = 0; j < 8; ++j) { minval[j] = 3.4e38f; minidx[j] = 0; }

  const __bf16* cbl_g = cb_bf + (size_t)level * K_CODES * D_DIM;
  const float*  nrm   = norms + level * K_CODES;
  const uint4*  gbase = (const uint4*)cbl_g;        // chunk stride = CHUNK_U4

#if RVQ_ASYNC
  rvq_stage_async(gbase, ldsCB, tid);               // prologue: chunk 0 -> buf0
#endif

  for (int nc = 0; nc < 32; ++nc) {                 // 32 chunks x 32 codes = 1024
    const int cur = nc & 1;
#if RVQ_ASYNC
    __syncthreads();   // all waves done reading buf[1-cur] (chunk nc-1)
    if (nc + 1 < 32) {
      rvq_stage_async(gbase + (size_t)(nc + 1) * CHUNK_U4,
                      ldsCB + (1 - cur) * CHUNK_U4, tid);
      // 6 outstanding = next chunk's; in-order completion => chunk nc landed
      __builtin_amdgcn_s_wait_asynccnt(6);
    } else {
      __builtin_amdgcn_s_wait_asynccnt(0);
    }
    __syncthreads();   // all waves' chunk-nc async writes visible
#else
    __syncthreads();
    const uint4* src = gbase + (size_t)nc * CHUNK_U4;
#pragma unroll
    for (int i = 0; i < 6; ++i)
      ldsCB[cur * CHUNK_U4 + tid + i * 256] = src[tid + i * 256];
    if (nc + 1 < 32)   // warm L2 for next chunk: emits global_prefetch_b8
      __builtin_prefetch((const void*)((const char*)src + 24576 + tid * 96), 0, 0);
    __syncthreads();
#endif

    const __bf16* cbl = (const __bf16*)(ldsCB + cur * CHUNK_U4);
#pragma unroll
    for (int sub = 0; sub < 2; ++sub) {
      const int n0 = nc * 32 + sub * 16;
      const __bf16* crow = cbl + (sub * 16 + mrow) * D_DIM;
      v8f acc = {};
#pragma unroll
      for (int c = 0; c < 12; ++c) {                // K = 384 in 12 steps of 32
        const int d0 = c * 32 + kbase;
        v8bf b0 = *(const v8bf*)(crow + d0);        // ds_load_b128
        v8bf b1 = *(const v8bf*)(crow + d0 + 16);   // ds_load_b128
        v16bf b;
#pragma unroll
        for (int i = 0; i < 8; ++i) { b[i] = b0[i]; b[i + 8] = b1[i]; }
        acc = __builtin_amdgcn_wmma_f32_16x16x32_bf16(
            false, a[c], false, b, (short)0, acc, false, false);
      }
      // score = ||e||^2 - 2 r.e  (||r||^2 dropped: constant per row for argmin)
      const float cbn  = nrm[n0 + mrow];
      const int   cand = n0 + mrow;
#pragma unroll
      for (int j = 0; j < 8; ++j) {
        float s = cbn - 2.f * acc[j];
        if (s < minval[j]) { minval[j] = s; minidx[j] = cand; }
      }
    }
  }

  // ---- argmin reduction across the 16 lanes of each half-wave
#pragma unroll
  for (int j = 0; j < 8; ++j) {
#pragma unroll
    for (int off = 8; off >= 1; off >>= 1) {
      float ov = __shfl_xor(minval[j], off, 16);
      int   oi = __shfl_xor(minidx[j], off, 16);
      if (ov < minval[j] || (ov == minval[j] && oi < minidx[j])) {
        minval[j] = ov; minidx[j] = oi;
      }
    }
  }

  // ---- emit indices (d_out is float-typed)
  float* outIdx = out + (size_t)B_N * D_DIM + (size_t)level * B_N;
  if (mrow == 0) {                                  // lanes 0 and 16
#pragma unroll
    for (int j = 0; j < 8; ++j) {
      int row = rowBase + (lane >> 4) * 8 + j;      // M=j (lane0) / M=j+8 (lane16)
      outIdx[row] = (float)minidx[j];
    }
  }

  // ---- gather codes (fp32), residual update in place, loss partial
  const float* cbl_f = cb + (size_t)level * K_CODES * D_DIM;
  float lsum = 0.f;
#pragma unroll
  for (int m = 0; m < 16; ++m) {
    int code = __shfl(minidx[m & 7], (m >> 3) * 16, 32);
    const int row = rowBase + m;
    const float* erow = cbl_f + (size_t)code * D_DIM;
    const float* rin  = resIn  + (size_t)row * D_DIM;
    float*       rout = resOut + (size_t)row * D_DIM;
#pragma unroll
    for (int t = 0; t < 12; ++t) {
      int d = lane + 32 * t;
      float nr = rin[d] - erow[d];
      rout[d] = nr;
      lsum += nr * nr;
    }
  }
#pragma unroll
  for (int off = 16; off >= 1; off >>= 1) lsum += __shfl_xor(lsum, off, 32);
  if (lane == 0) atomicAdd(lossAcc, lsum);
}

// ---------------------------------------------------------------------------
// Final: quant_sum = z - residual_final; scale the loss accumulator
// ---------------------------------------------------------------------------
__global__ __launch_bounds__(256) void rvq_final_kernel(
    const float* __restrict__ z, const float* __restrict__ res,
    float* __restrict__ out, const float* __restrict__ lossAcc) {
  size_t i = (size_t)blockIdx.x * 256 + threadIdx.x;
  out[i] = z[i] - res[i];
  if (i == 0) {
    out[(size_t)B_N * D_DIM + (size_t)L_LEV * B_N] =
        (*lossAcc) * (1.f + BETA_C) / ((float)B_N * (float)D_DIM);
  }
}

// ---------------------------------------------------------------------------
extern "C" void kernel_launch(void* const* d_in, const int* in_sizes, int n_in,
                              void* d_out, int out_size, void* d_ws, size_t ws_size,
                              hipStream_t stream) {
  const float* z  = (const float*)d_in[0];               // [B, D]
  const float* cb = (const float*)d_in[1];               // [L, K, D]
  float* out = (float*)d_out;

  char* ws = (char*)d_ws;
  float*  res   = (float*)ws;                                            // B*D f32
  __bf16* cb_bf = (__bf16*)(ws + (size_t)B_N * D_DIM * 4);               // L*K*D bf16
  float*  norms = (float*)(ws + (size_t)B_N * D_DIM * 4
                              + (size_t)L_LEV * K_CODES * D_DIM * 2);    // L*K f32
  float*  lossAcc = norms + (size_t)L_LEV * K_CODES;                     // 1 f32

  rvq_prep_kernel<<<(L_LEV * K_CODES) / 8, 256, 0, stream>>>(cb, cb_bf, norms, lossAcc);

  for (int level = 0; level < L_LEV; ++level) {
    const float* rin = (level == 0) ? z : res;
    rvq_level_kernel<<<B_N / 128, 256, 0, stream>>>(
        rin, res, cb, cb_bf, norms, out, lossAcc, level);
  }

  rvq_final_kernel<<<(B_N * (size_t)D_DIM) / 256, 256, 0, stream>>>(z, res, out, lossAcc);
}
